// HyperGraphSparseAttention_20985210208482
// MI455X (gfx1250) — matrix operand order repaired and checked by reading, not compiled, not run on hardware
//
#include <hip/hip_runtime.h>
#include <hip/hip_bf16.h>

// ---------------- problem constants ----------------
#define NSEQ   2048
#define EMBED  1024
#define NH     16      // query heads
#define NKV    4       // kv heads
#define GQ     (NH / NKV)
#define HD     64      // head dim
#define KN     4       // hyper nodes
#define SCALE  0.125f  // 64^-0.5
#define NEGBIG (-1.0e9f)

typedef __bf16 v16bf __attribute__((ext_vector_type(16)));
typedef float  v8f   __attribute__((ext_vector_type(8)));
typedef int    v4i   __attribute__((vector_size(16)));   // matches async-LDS builtin param

#if __has_builtin(__builtin_amdgcn_global_load_async_to_lds_b128)
#define HAVE_ASYNC_LDS 1
#else
#define HAVE_ASYNC_LDS 0
#endif

__device__ __forceinline__ void wait_async_lds() {
#if HAVE_ASYNC_LDS
#if __has_builtin(__builtin_amdgcn_s_wait_asynccnt)
    __builtin_amdgcn_s_wait_asynccnt(0);
#else
    asm volatile("s_wait_asynccnt 0x0" ::: "memory");
#endif
#endif
}

// A-fragment (16x32 bf16) K index for element e, lane-half hl (ISA 05_wmma 7.12.2)
__device__ __forceinline__ int kmapA(int e, int hl) {
    return (e < 8) ? (hl * 8 + e) : (16 + hl * 8 + (e - 8));
}

// ---------------- f32 -> bf16 convert ----------------
__global__ void cvt_f32_bf16(const float* __restrict__ in, __bf16* __restrict__ out, int n) {
    int i = blockIdx.x * blockDim.x + threadIdx.x;
    if (i < n) out[i] = (__bf16)in[i];
}

// ---------------- generic bf16 WMMA GEMM: C[MxN] = A[MxK] @ B[KxN], row major ----------------
// grid = (N/64, M/64), block = 128 (4 waves). Requires M%64==0, N%64==0, K%32==0.
__global__ void gemm_bf16(const __bf16* __restrict__ A, const __bf16* __restrict__ B,
                          float* __restrict__ C, int M, int K, int N) {
    // pitch 48 halves = 96B: multiple of 16B (aligned b128 reads), 24 dword-banks (spread)
    __shared__ __align__(16) __bf16 As[64][48];    // A tile, row-major, k contiguous
    __shared__ __align__(16) __bf16 BsT[64][48];   // B tile TRANSPOSED: [col][k]

    const int tid  = threadIdx.x;
    const int lane = tid & 31;
    const int w    = tid >> 5;
    const int hl   = lane >> 4;
    const int l16  = lane & 15;

    const int rowBase = blockIdx.y * 64;
    const int colBase = blockIdx.x * 64;
    const int wr = (w >> 1) * 32;
    const int wc = (w & 1) * 32;

    v8f acc[2][2] = {};

    for (int kk = 0; kk < K; kk += 32) {
        // stage A tile 64x32 (async DMA to LDS when available: 256 x 16B chunks)
#if HAVE_ASYNC_LDS
        for (int i = tid; i < 256; i += 128) {
            int r = i >> 2, ch = (i & 3) * 8;
            __builtin_amdgcn_global_load_async_to_lds_b128(
                (v4i*)(const_cast<__bf16*>(A + (size_t)(rowBase + r) * K + kk + ch)),
                (v4i*)&As[r][ch], 0, 0);
        }
#else
        for (int i = tid; i < 64 * 32; i += 128) {
            int r = i >> 5, c = i & 31;
            As[r][c] = A[(size_t)(rowBase + r) * K + kk + c];
        }
#endif
        // stage B tile 32x64 transposed: BsT[col][k]
        for (int i = tid; i < 32 * 64; i += 128) {
            int r = i >> 6, c = i & 63;
            BsT[c][r] = B[(size_t)(kk + r) * N + colBase + c];
        }
        if (kk + 32 < K) __builtin_prefetch(&A[(size_t)(rowBase + (tid >> 1)) * K + kk + 32], 0, 0);
        wait_async_lds();
        __syncthreads();

        v16bf afr[2], bfr[2];
        for (int mi = 0; mi < 2; ++mi) {
            int m = wr + mi * 16 + l16;
            for (int e = 0; e < 16; ++e) afr[mi][e] = As[m][kmapA(e, hl)];
        }
        for (int ni = 0; ni < 2; ++ni) {
            int c = wc + ni * 16 + l16;
            for (int e = 0; e < 16; ++e) bfr[ni][e] = BsT[c][kmapA(e, hl)];
        }
        for (int mi = 0; mi < 2; ++mi)
            for (int ni = 0; ni < 2; ++ni)
                acc[mi][ni] = __builtin_amdgcn_wmma_f32_16x16x32_bf16(
                    false, afr[mi], false, bfr[ni], (short)0, acc[mi][ni], false, false);
        __syncthreads();
    }

    for (int mi = 0; mi < 2; ++mi)
        for (int ni = 0; ni < 2; ++ni)
            for (int v = 0; v < 8; ++v) {
                int m = rowBase + wr + mi * 16 + hl * 8 + v;
                int c = colBase + wc + ni * 16 + l16;
                C[(size_t)m * N + c] = acc[mi][ni][v];
            }
}

// ---------------- routing ----------------
__global__ void route_scan(const float* __restrict__ logits, int* __restrict__ member,
                           int* __restrict__ pos) {
    int tid = threadIdx.x;
    if (tid >= NH * KN) return;
    int h = tid >> 2, n = tid & 3;
    int cnt = 0;
    for (int t = 0; t < NSEQ; ++t) {
        const float* l = logits + (size_t)t * (NH * KN) + h * KN;
        float lv[4] = {l[0], l[1], l[2], l[3]};
        int i0 = 0;
        for (int i = 1; i < 4; ++i) if (lv[i] > lv[i0]) i0 = i;
        int i1 = (i0 == 0) ? 1 : 0;
        for (int i = 0; i < 4; ++i) if (i != i0 && lv[i] > lv[i1]) i1 = i;
        int mem = (n == i0 || n == i1) ? 1 : 0;
        size_t idx = (size_t)(h * KN + n) * NSEQ + t;
        member[idx] = mem;
        pos[idx] = cnt;
        cnt += mem;
    }
}

__global__ void topk_weights(const float* __restrict__ logits, int* __restrict__ tki,
                             float* __restrict__ tkw) {
    int idx = blockIdx.x * blockDim.x + threadIdx.x;
    if (idx >= NSEQ * NH) return;
    int t = idx / NH, h = idx % NH;
    const float* l = logits + (size_t)t * (NH * KN) + h * KN;
    float lv[4] = {l[0], l[1], l[2], l[3]};
    int i0 = 0;
    for (int i = 1; i < 4; ++i) if (lv[i] > lv[i0]) i0 = i;
    int i1 = (i0 == 0) ? 1 : 0;
    for (int i = 0; i < 4; ++i) if (i != i0 && lv[i] > lv[i1]) i1 = i;
    float w0 = 1.0f / (1.0f + __expf(lv[i1] - lv[i0]));
    tki[idx * 2 + 0] = i0;
    tki[idx * 2 + 1] = i1;
    tkw[idx * 2 + 0] = w0;
    tkw[idx * 2 + 1] = 1.0f - w0;
}

// ---------------- flash attention per (head, node) with per-timeline RoPE ----------------
// grid = (NSEQ/64, NH*KN), block = 128 (4 waves, each owns 16 query rows)
// outs layout: [((h*NSEQ + t)*KN + n)*HD + d]
__global__ void attn_kernel(const float* __restrict__ Q, const float* __restrict__ Kp,
                            const float* __restrict__ Vp, const int* __restrict__ member,
                            const int* __restrict__ pos, float* __restrict__ outs) {
    __shared__ float InvF[32];
    __shared__ __align__(16) __bf16 Qr[64][72];     // rotated+scaled Q, row-major [row][d]
    __shared__ __align__(16) __bf16 Krs[16][72];    // rotated K, row-major [key][d]
    __shared__ __align__(16) __bf16 Vt[64][48];     // V transposed [d][key], keys 16..31 = 0
    __shared__ __align__(16) __bf16 PwB[4][16][48]; // P per wave [row][key], cols 16..31 = 0

    const int tid  = threadIdx.x;
    const int lane = tid & 31;
    const int w    = tid >> 5;
    const int hl   = lane >> 4;
    const int l16  = lane & 15;

    const int h  = blockIdx.y / KN;
    const int n  = blockIdx.y % KN;
    const int qb = blockIdx.x;
    const int kvh = h / GQ;
    const int* memb = member + (size_t)(h * KN + n) * NSEQ;
    const int* posn = pos + (size_t)(h * KN + n) * NSEQ;

    if (tid < 32) InvF[tid] = __powf(10000.0f, -((float)(2 * tid)) / 64.0f);
    // zero pad regions read by zero-padded fragments
    for (int i = tid; i < 64 * 16; i += 128) Vt[i >> 4][16 + (i & 15)] = (__bf16)0.0f;
    for (int i = tid; i < 4 * 16 * 16; i += 128) {
        int ww = i >> 8, r = (i >> 4) & 15, c = i & 15;
        PwB[ww][r][16 + c] = (__bf16)0.0f;
    }
    __syncthreads();  // InvF ready

    // cooperative RoPE+scale of the 64-row Q tile into bf16 LDS (one sincos per pair)
    for (int i = tid; i < 64 * 32; i += 128) {
        int r = i >> 5, fi = i & 31;
        int qi = qb * 64 + r;
        float p = (float)posn[qi];
        float sv, cv;
        __sincosf(p * InvF[fi], &sv, &cv);
        const float* qrow = Q + (size_t)qi * (NH * HD) + h * HD;
        float a = qrow[fi], b = qrow[fi + 32];
        Qr[r][fi]      = (__bf16)((a * cv - b * sv) * SCALE);
        Qr[r][fi + 32] = (__bf16)((b * cv + a * sv) * SCALE);
    }
    __syncthreads();

    // per-wave Q A-fragments (rows w*16 .. +15)
    const int mrow = w * 16 + l16;
    v16bf aq[2];
    for (int f = 0; f < 2; ++f)
        for (int e = 0; e < 16; ++e)
            aq[f][e] = Qr[mrow][kmapA(e, hl) + 32 * f];

    float mrun[8], lsum[8];
    for (int v = 0; v < 8; ++v) { mrun[v] = -3.0e38f; lsum[v] = 0.0f; }
    v8f oacc[4] = {};

    const int jmax = qb * 4 + 3;
    for (int j = 0; j <= jmax; ++j) {
        __syncthreads();  // previous-iteration readers done
        // stage rotated K (bf16) and transposed V (bf16)
        for (int i = tid; i < 16 * 32; i += 128) {
            int r = i >> 5, fi = i & 31;
            int kg = j * 16 + r;
            float p = (float)posn[kg];
            float sv, cv;
            __sincosf(p * InvF[fi], &sv, &cv);
            const float* krow = Kp + (size_t)kg * (NKV * HD) + kvh * HD;
            float a = krow[fi], b = krow[fi + 32];
            Krs[r][fi]      = (__bf16)(a * cv - b * sv);
            Krs[r][fi + 32] = (__bf16)(b * cv + a * sv);
        }
        for (int i = tid; i < 16 * 64; i += 128) {
            int r = i >> 6, c = i & 63;  // r = key, c = dim
            Vt[c][r] = (__bf16)Vp[(size_t)(j * 16 + r) * (NKV * HD) + kvh * HD + c];
        }
        __syncthreads();

        // score B-fragments: B[k=d][n=key]
        v16bf bk[2];
        for (int f = 0; f < 2; ++f)
            for (int e = 0; e < 16; ++e)
                bk[f][e] = Krs[l16][kmapA(e, hl) + 32 * f];

        v8f sc = {};
        sc = __builtin_amdgcn_wmma_f32_16x16x32_bf16(false, aq[0], false, bk[0], (short)0, sc, false, false);
        sc = __builtin_amdgcn_wmma_f32_16x16x32_bf16(false, aq[1], false, bk[1], (short)0, sc, false, false);

        // mask (causal + membership) and online softmax
        const int kg_lane = j * 16 + l16;
        const int mem_lane = memb[kg_lane];
        for (int v = 0; v < 8; ++v) {
            int qg = qb * 64 + w * 16 + hl * 8 + v;
            float sval = (mem_lane && kg_lane <= qg) ? sc[v] : NEGBIG;
            float mj = sval;
            for (int off = 1; off < 16; off <<= 1) mj = fmaxf(mj, __shfl_xor(mj, off, 32));
            float mnew = fmaxf(mrun[v], mj);
            float alpha = __expf(mrun[v] - mnew);
            float pval = __expf(sval - mnew);
            float psum = pval;
            for (int off = 1; off < 16; off <<= 1) psum += __shfl_xor(psum, off, 32);
            lsum[v] = lsum[v] * alpha + psum;
            mrun[v] = mnew;
            for (int dt = 0; dt < 4; ++dt) oacc[dt][v] *= alpha;
            PwB[w][hl * 8 + v][l16] = (__bf16)pval;  // C-layout -> LDS (bf16)
        }
        // same-wave LDS RAW fence before re-reading P in A-layout
        asm volatile("s_wait_dscnt 0x0" ::: "memory");

        // P A-fragment: kmapA sends e>=8 into the pre-zeroed 16..31 region
        v16bf ap;
        for (int e = 0; e < 16; ++e) ap[e] = PwB[w][l16][kmapA(e, hl)];
        // V B-fragments: B[k=key][n=dim]; keys 16..31 pre-zeroed
        for (int dt = 0; dt < 4; ++dt) {
            v16bf bv;
            for (int e = 0; e < 16; ++e) bv[e] = Vt[dt * 16 + l16][kmapA(e, hl)];
            oacc[dt] = __builtin_amdgcn_wmma_f32_16x16x32_bf16(false, ap, false, bv, (short)0, oacc[dt], false, false);
        }
    }

    for (int dt = 0; dt < 4; ++dt)
        for (int v = 0; v < 8; ++v) {
            int qg = qb * 64 + w * 16 + hl * 8 + v;
            int d = dt * 16 + l16;
            outs[(((size_t)h * NSEQ + qg) * KN + n) * HD + d] = oacc[dt][v] / lsum[v];
        }
}

// ---------------- combine by routing weights ----------------
__global__ void combine_kernel(const float* __restrict__ outs, const int* __restrict__ tki,
                               const float* __restrict__ tkw, float* __restrict__ Y) {
    int idx = blockIdx.x * blockDim.x + threadIdx.x;
    if (idx >= NSEQ * NH * HD) return;
    int t = idx / (NH * HD);
    int r = idx % (NH * HD);
    int h = r / HD, d = r % HD;
    int base = (t * NH + h) * 2;
    int i0 = tki[base], i1 = tki[base + 1];
    float w0 = tkw[base], w1 = tkw[base + 1];
    size_t ob = ((size_t)h * NSEQ + t) * KN;
    Y[(size_t)t * (NH * HD) + h * HD + d] =
        w0 * outs[(ob + i0) * HD + d] + w1 * outs[(ob + i1) * HD + d];
}

// ---------------- launcher ----------------
extern "C" void kernel_launch(void* const* d_in, const int* in_sizes, int n_in,
                              void* d_out, int out_size, void* d_ws, size_t ws_size,
                              hipStream_t stream) {
    const float* x  = (const float*)d_in[0];
    const float* wq = (const float*)d_in[1];
    const float* wk = (const float*)d_in[2];
    const float* wv = (const float*)d_in[3];
    const float* wr = (const float*)d_in[4];
    const float* wo = (const float*)d_in[5];
    float* out = (float*)d_out;

    char* ws = (char*)d_ws;
    size_t off = 0;
    auto alloc = [&](size_t bytes) { char* p = ws + off; off += (bytes + 255) & ~(size_t)255; return p; };

    __bf16* xb  = (__bf16*)alloc((size_t)NSEQ * EMBED * 2);
    __bf16* wqb = (__bf16*)alloc((size_t)EMBED * NH * HD * 2);
    __bf16* wkb = (__bf16*)alloc((size_t)EMBED * NKV * HD * 2);
    __bf16* wvb = (__bf16*)alloc((size_t)EMBED * NKV * HD * 2);
    __bf16* wrb = (__bf16*)alloc((size_t)EMBED * NH * KN * 2);
    __bf16* wob = (__bf16*)alloc((size_t)NH * HD * EMBED * 2);
    float*  Qp  = (float*)alloc((size_t)NSEQ * NH * HD * 4);
    float*  Kp  = (float*)alloc((size_t)NSEQ * NKV * HD * 4);
    float*  Vp  = (float*)alloc((size_t)NSEQ * NKV * HD * 4);
    float*  Lg  = (float*)alloc((size_t)NSEQ * NH * KN * 4);
    int*    tki = (int*)alloc((size_t)NSEQ * NH * 2 * 4);
    float*  tkw = (float*)alloc((size_t)NSEQ * NH * 2 * 4);
    int*    mem = (int*)alloc((size_t)NH * KN * NSEQ * 4);
    int*    pos = (int*)alloc((size_t)NH * KN * NSEQ * 4);
    float*  outs = (float*)alloc((size_t)NH * NSEQ * KN * HD * 4);
    float*  Y   = (float*)alloc((size_t)NSEQ * NH * HD * 4);
    __bf16* Yb  = (__bf16*)alloc((size_t)NSEQ * NH * HD * 2);

    auto cvt = [&](const float* in, __bf16* o, int n) {
        cvt_f32_bf16<<<(n + 255) / 256, 256, 0, stream>>>(in, o, n);
    };
    cvt(x,  xb,  NSEQ * EMBED);
    cvt(wq, wqb, EMBED * NH * HD);
    cvt(wk, wkb, EMBED * NKV * HD);
    cvt(wv, wvb, EMBED * NKV * HD);
    cvt(wr, wrb, EMBED * NH * KN);
    cvt(wo, wob, NH * HD * EMBED);

    // projections
    gemm_bf16<<<dim3((NH * HD) / 64, NSEQ / 64), 128, 0, stream>>>(xb, wqb, Qp, NSEQ, EMBED, NH * HD);
    gemm_bf16<<<dim3((NKV * HD) / 64, NSEQ / 64), 128, 0, stream>>>(xb, wkb, Kp, NSEQ, EMBED, NKV * HD);
    gemm_bf16<<<dim3((NKV * HD) / 64, NSEQ / 64), 128, 0, stream>>>(xb, wvb, Vp, NSEQ, EMBED, NKV * HD);
    gemm_bf16<<<dim3((NH * KN) / 64, NSEQ / 64), 128, 0, stream>>>(xb, wrb, Lg, NSEQ, EMBED, NH * KN);

    // routing
    route_scan<<<1, 64, 0, stream>>>(Lg, mem, pos);
    topk_weights<<<(NSEQ * NH + 255) / 256, 256, 0, stream>>>(Lg, tki, tkw);

    // per-(head,node) flash attention
    attn_kernel<<<dim3(NSEQ / 64, NH * KN), 128, 0, stream>>>(Qp, Kp, Vp, mem, pos, outs);

    // combine + output projection
    combine_kernel<<<(NSEQ * NH * HD + 255) / 256, 256, 0, stream>>>(outs, tki, tkw, Y);
    cvt(Y, Yb, NSEQ * NH * HD);
    gemm_bf16<<<dim3(EMBED / 64, NSEQ / 64), 128, 0, stream>>>(Yb, wob, out, NSEQ, NH * HD, EMBED);
}
